// GraphSage_73589969650012
// MI455X (gfx1250) — matrix operand order repaired and checked by reading, not compiled, not run on hardware
//
#include <hip/hip_runtime.h>

#define EPSV 1e-5f

typedef __attribute__((ext_vector_type(16))) _Float16 v16h;
typedef __attribute__((ext_vector_type(8)))  _Float16 v8h;
typedef __attribute__((ext_vector_type(8)))  float    v8f;

// ----------------------------------------------------------------------------
// utility fills / converts (grid-stride)
// ----------------------------------------------------------------------------
__global__ void k_fill_f32(float* p, float v, size_t n){
  size_t i = (size_t)blockIdx.x*blockDim.x + threadIdx.x;
  size_t st = (size_t)gridDim.x*blockDim.x;
  for (; i < n; i += st) p[i] = v;
}
__global__ void k_fill_f16(_Float16* p, float v, size_t n){
  size_t i = (size_t)blockIdx.x*blockDim.x + threadIdx.x;
  size_t st = (size_t)gridDim.x*blockDim.x;
  _Float16 h = (_Float16)v;
  for (; i < n; i += st) p[i] = h;
}
__global__ void k_fill_i32(int* p, int v, size_t n){
  size_t i = (size_t)blockIdx.x*blockDim.x + threadIdx.x;
  size_t st = (size_t)gridDim.x*blockDim.x;
  for (; i < n; i += st) p[i] = v;
}
__global__ void k_fill_u32(unsigned* p, unsigned v, size_t n){
  size_t i = (size_t)blockIdx.x*blockDim.x + threadIdx.x;
  size_t st = (size_t)gridDim.x*blockDim.x;
  for (; i < n; i += st) p[i] = v;
}
__global__ void k_cvt_f16(const float* __restrict__ in, _Float16* __restrict__ out, size_t n){
  size_t i = (size_t)blockIdx.x*blockDim.x + threadIdx.x;
  size_t st = (size_t)gridDim.x*blockDim.x;
  for (; i < n; i += st) out[i] = (_Float16)in[i];
}

// ----------------------------------------------------------------------------
// WMMA GEMM: C[M,N] = A[M,K] (f16, row-major) * B[N,K]^T (f16, row-major) + bias
// one wave32 per 32x32 tile, 2x2 fragments of v_wmma_f32_16x16x32_f16
// M (gridDim.y*32) must be padded to a multiple of 32; K multiple of 32.
// ----------------------------------------------------------------------------
__global__ __launch_bounds__(32) void k_wmma_gemm_nt(
    const _Float16* __restrict__ A, const _Float16* __restrict__ B,
    const float* __restrict__ bias, float* __restrict__ C,
    int N, int K)
{
  const int lane = threadIdx.x & 31;
  const int hf   = lane >> 4;     // 0 or 1
  const int l16  = lane & 15;
  const int tn   = blockIdx.x;    // 32-wide col tile
  const int tm   = blockIdx.y;    // 32-tall row tile

  // A fragment (16-bit 16x32): lanes 0-15 take K = [k, k+16);
  // lanes 16-31 take K = [k+16, k+32).
  const _Float16* a0 = A + (size_t)(tm*32 + l16) * K + hf*16;
  const _Float16* a1 = a0 + (size_t)16 * K;
  // B fragment (16-bit 32x16, K-striped): lane L<16 -> col N=L, K=[k,k+16);
  // lanes 16-31 -> K=[k+16,k+32).  Contiguous 32-byte load from row-major W.
  const _Float16* b0 = B + (size_t)(tn*32 + l16) * K + hf*16;
  const _Float16* b1 = b0 + (size_t)16 * K;

  v8f acc00 = {}; v8f acc01 = {}; v8f acc10 = {}; v8f acc11 = {};

  for (int k = 0; k < K; k += 32){
    union { v16h v; v8h h[2]; } A0, A1;
    A0.h[0] = *(const v8h*)(a0 + k);
    A0.h[1] = *(const v8h*)(a0 + k + 8);
    A1.h[0] = *(const v8h*)(a1 + k);
    A1.h[1] = *(const v8h*)(a1 + k + 8);
    v16h B0 = *(const v16h*)(b0 + k);
    v16h B1 = *(const v16h*)(b1 + k);
    __builtin_prefetch(a0 + k + 128, 0, 1);   // global_prefetch_b8
    __builtin_prefetch(b0 + k + 128, 0, 1);
    acc00 = __builtin_amdgcn_wmma_f32_16x16x32_f16(false, A0.v, false, B0, (short)0, acc00, false, false);
    acc01 = __builtin_amdgcn_wmma_f32_16x16x32_f16(false, A0.v, false, B1, (short)0, acc01, false, false);
    acc10 = __builtin_amdgcn_wmma_f32_16x16x32_f16(false, A1.v, false, B0, (short)0, acc10, false, false);
    acc11 = __builtin_amdgcn_wmma_f32_16x16x32_f16(false, A1.v, false, B1, (short)0, acc11, false, false);
  }

  const float bs0 = bias[tn*32 + l16];
  const float bs1 = bias[tn*32 + 16 + l16];
  const size_t r0 = (size_t)tm*32 + hf*8;     // C layout: VGPR r <-> M = r + 8*hf
  const size_t c0 = (size_t)tn*32 + l16;
  for (int r = 0; r < 8; ++r){
    C[(r0 + r)      * N + c0]      = acc00[r] + bs0;
    C[(r0 + r)      * N + c0 + 16] = acc01[r] + bs1;
    C[(r0 + 16 + r) * N + c0]      = acc10[r] + bs0;
    C[(r0 + 16 + r) * N + c0 + 16] = acc11[r] + bs1;
  }
}

// ----------------------------------------------------------------------------
// column mean / population variance, one block per column
// ----------------------------------------------------------------------------
__global__ void k_colstats(const float* __restrict__ H, int M, int N,
                           float* __restrict__ mean, float* __restrict__ var){
  __shared__ float s1[256];
  __shared__ float s2[256];
  const int c = blockIdx.x;
  float a = 0.f, b = 0.f;
  for (int r = threadIdx.x; r < M; r += blockDim.x){
    float v = H[(size_t)r*N + c];
    a += v; b += v*v;
  }
  s1[threadIdx.x] = a; s2[threadIdx.x] = b;
  __syncthreads();
  for (int s = blockDim.x >> 1; s > 0; s >>= 1){
    if ((int)threadIdx.x < s){ s1[threadIdx.x] += s1[threadIdx.x+s]; s2[threadIdx.x] += s2[threadIdx.x+s]; }
    __syncthreads();
  }
  if (threadIdx.x == 0){
    float m = s1[0] / (float)M;
    mean[c] = m;
    var[c]  = s2[0] / (float)M - m*m;
  }
}

__global__ void k_bn_relu(float* __restrict__ H, const float* __restrict__ mean,
                          const float* __restrict__ var, const float* __restrict__ g,
                          const float* __restrict__ be, size_t total, int N){
  size_t i = (size_t)blockIdx.x*blockDim.x + threadIdx.x;
  size_t st = (size_t)gridDim.x*blockDim.x;
  for (; i < total; i += st){
    int c = (int)(i % (size_t)N);
    float s = g[c] * rsqrtf(var[c] + EPSV);
    float v = (H[i] - mean[c]) * s + be[c];
    H[i] = v > 0.f ? v : 0.f;
  }
}

// ----------------------------------------------------------------------------
// per-node projections p1 = h . Wp[:F], p2 = h . Wp[F:]; one block per node
// ----------------------------------------------------------------------------
__global__ void k_node_proj(const float* __restrict__ H, const float* __restrict__ Wp,
                            float* __restrict__ p1, float* __restrict__ p2, int F){
  __shared__ float s1[256];
  __shared__ float s2[256];
  const int i = blockIdx.x;
  float a = 0.f, b = 0.f;
  for (int f = threadIdx.x; f < F; f += blockDim.x){
    float h = H[(size_t)i*F + f];
    a += h * Wp[f];
    b += h * Wp[F + f];
  }
  s1[threadIdx.x] = a; s2[threadIdx.x] = b;
  __syncthreads();
  for (int s = blockDim.x >> 1; s > 0; s >>= 1){
    if ((int)threadIdx.x < s){ s1[threadIdx.x] += s1[threadIdx.x+s]; s2[threadIdx.x] += s2[threadIdx.x+s]; }
    __syncthreads();
  }
  if (threadIdx.x == 0){ p1[i] = s1[0]; p2[i] = s2[0]; }
}

// ----------------------------------------------------------------------------
// segment softmax over dst + score
// ----------------------------------------------------------------------------
__device__ inline unsigned f32key(float f){
  unsigned u = __float_as_uint(f);
  return (u & 0x80000000u) ? ~u : (u | 0x80000000u);
}
__device__ inline float keyf32(unsigned k){
  unsigned u = (k & 0x80000000u) ? (k ^ 0x80000000u) : ~k;
  return __uint_as_float(u);
}

__global__ void k_edge_e(const float* __restrict__ p1, const float* __restrict__ p2,
                         const int* __restrict__ src, const int* __restrict__ dst,
                         const float* __restrict__ bp, float* __restrict__ e,
                         unsigned* __restrict__ mkey, int E){
  int j = blockIdx.x*blockDim.x + threadIdx.x;
  if (j >= E) return;
  float v = p1[src[j]] + p2[dst[j]] + bp[0];
  e[j] = v;
  atomicMax(&mkey[dst[j]], f32key(v));
}
__global__ void k_edge_ex(const float* __restrict__ e, const unsigned* __restrict__ mkey,
                          const int* __restrict__ dst, float* __restrict__ ex,
                          float* __restrict__ den, int E){
  int j = blockIdx.x*blockDim.x + threadIdx.x;
  if (j >= E) return;
  float m = keyf32(mkey[dst[j]]);
  float x = expf(e[j] - m);
  ex[j] = x;
  atomicAdd(&den[dst[j]], x);
}
__global__ void k_edge_score(const float* __restrict__ ex, const float* __restrict__ den,
                             const int* __restrict__ dst, float* __restrict__ sc, int E){
  int j = blockIdx.x*blockDim.x + threadIdx.x;
  if (j >= E) return;
  sc[j] = ex[j] / den[dst[j]] + 0.5f;
}

// ----------------------------------------------------------------------------
// stable argsort by descending score: key = sortable(-score) << 32 | index
// ----------------------------------------------------------------------------
__global__ void k_build_keys(const float* __restrict__ sc, unsigned long long* __restrict__ keys,
                             int E, int npad){
  int i = blockIdx.x*blockDim.x + threadIdx.x;
  if (i >= npad) return;
  if (i < E){
    unsigned k = f32key(-sc[i]);
    keys[i] = ((unsigned long long)k << 32) | (unsigned)i;
  } else {
    keys[i] = 0xFFFFFFFFFFFFFFFFULL;
  }
}
__global__ void k_bitonic(unsigned long long* __restrict__ keys, int j, int k){
  int i = blockIdx.x*blockDim.x + threadIdx.x;
  int ixj = i ^ j;
  if (ixj > i){
    unsigned long long a = keys[i], b = keys[ixj];
    bool up = ((i & k) == 0);
    if ((a > b) == up){ keys[i] = b; keys[ixj] = a; }
  }
}

// ----------------------------------------------------------------------------
// greedy matching — inherently sequential scan (single thread)
// ----------------------------------------------------------------------------
__global__ void k_match(const unsigned long long* __restrict__ keys,
                        const int* __restrict__ src, const int* __restrict__ dst,
                        const float* __restrict__ sc,
                        int* __restrict__ cluster, float* __restrict__ mult,
                        int E, int Nn){
  if (blockIdx.x != 0 || threadIdx.x != 0) return;
  int ec = 0;
  for (int j = 0; j < E; ++j){
    int idx = (int)(keys[j] & 0xFFFFFFFFULL);
    int s = src[idx], d = dst[idx];
    if (cluster[s] < 0 && cluster[d] < 0){
      cluster[s] = ec;
      cluster[d] = ec;
      mult[ec] = sc[idx];
      ++ec;
    }
  }
  int rank = 0;
  for (int i = 0; i < Nn; ++i){
    if (cluster[i] < 0) cluster[i] = ec + rank++;
  }
}

__global__ void k_scatter_batch(const int* __restrict__ cluster, const int* __restrict__ batch,
                                int* __restrict__ nb, int Nn){
  int i = blockIdx.x*blockDim.x + threadIdx.x;
  if (i < Nn) nb[cluster[i]] = batch[i];
}

__global__ void k_scatter_add(const float* __restrict__ H, const int* __restrict__ cluster,
                              float* __restrict__ out, size_t total, int F){
  size_t i = (size_t)blockIdx.x*blockDim.x + threadIdx.x;
  size_t st = (size_t)gridDim.x*blockDim.x;
  for (; i < total; i += st){
    size_t row = i / (size_t)F;
    int f = (int)(i % (size_t)F);
    atomicAdd(&out[(size_t)cluster[row]*F + f], H[i]);
  }
}

__global__ void k_scale_cvt(const float* __restrict__ nx, const float* __restrict__ mult,
                            _Float16* __restrict__ out, size_t total, int F){
  size_t i = (size_t)blockIdx.x*blockDim.x + threadIdx.x;
  size_t st = (size_t)gridDim.x*blockDim.x;
  for (; i < total; i += st){
    size_t row = i / (size_t)F;
    out[i] = (_Float16)(nx[i] * mult[row]);
  }
}

// ----------------------------------------------------------------------------
// batch mean pooling + small FC layers
// ----------------------------------------------------------------------------
__global__ void k_pool(const float* __restrict__ H, const int* __restrict__ nb,
                       float* __restrict__ sums, float* __restrict__ cnt,
                       size_t total, int F){
  size_t i = (size_t)blockIdx.x*blockDim.x + threadIdx.x;
  size_t st = (size_t)gridDim.x*blockDim.x;
  for (; i < total; i += st){
    size_t row = i / (size_t)F;
    int f = (int)(i % (size_t)F);
    int b = nb[row];
    atomicAdd(&sums[(size_t)b*F + f], H[i]);
    if (f == 0) atomicAdd(&cnt[b], 1.0f);
  }
}
__global__ void k_pooled(const float* __restrict__ sums, const float* __restrict__ cnt,
                         float* __restrict__ pooled, int Bb, int F){
  int i = blockIdx.x*blockDim.x + threadIdx.x;
  if (i >= Bb*F) return;
  int b = i / F;
  pooled[i] = sums[i] / fmaxf(cnt[b], 1.0f);
}
__global__ void k_fc(const float* __restrict__ X, const float* __restrict__ W,
                     const float* __restrict__ b, float* __restrict__ Y,
                     int M, int N, int K, int dorelu){
  int idx = blockIdx.x*blockDim.x + threadIdx.x;
  if (idx >= M*N) return;
  int m = idx / N, n = idx % N;
  const float* xr = X + (size_t)m*K;
  const float* wr = W + (size_t)n*K;
  float s = b[n];
  for (int k = 0; k < K; ++k) s += xr[k]*wr[k];
  Y[idx] = dorelu ? fmaxf(s, 0.f) : s;
}

// ----------------------------------------------------------------------------
extern "C" void kernel_launch(void* const* d_in, const int* in_sizes, int n_in,
                              void* d_out, int out_size, void* d_ws, size_t ws_size,
                              hipStream_t stream) {
  (void)in_sizes; (void)n_in; (void)out_size; (void)ws_size;

  const int Nn = 50000, Ee = 400000, Bb = 50, OUTC = 10;
  const int F0 = 512, H1n = 1024, H2n = 512, FC1 = 200;
  const int Mpad = 50016;         // 50000 rounded up to multiple of 32
  const int NPAD = 1 << 19;       // 524288 >= E, power of two for bitonic sort

  const float* x    = (const float*)d_in[0];
  const int*   ei   = (const int*)  d_in[1];
  const int*   batch= (const int*)  d_in[2];
  const float* W1   = (const float*)d_in[3];
  const float* b1   = (const float*)d_in[4];
  const float* g1v  = (const float*)d_in[5];
  const float* be1  = (const float*)d_in[6];
  const float* Wp   = (const float*)d_in[7];
  const float* bp   = (const float*)d_in[8];
  const float* W2   = (const float*)d_in[9];
  const float* b2   = (const float*)d_in[10];
  const float* g2v  = (const float*)d_in[11];
  const float* be2  = (const float*)d_in[12];
  const float* Wfc  = (const float*)d_in[13];
  const float* bfc  = (const float*)d_in[14];
  const float* Wfc1 = (const float*)d_in[15];
  const float* bfc1 = (const float*)d_in[16];
  const int* src = ei;
  const int* dst = ei + Ee;
  float* out = (float*)d_out;

  // ---- carve workspace (aligned 256B) ----
  char* base = (char*)d_ws;
  size_t off = 0;
  auto carve = [&](size_t bytes) -> char* {
    char* p = base + off;
    off = (off + bytes + 255) & ~(size_t)255;
    return p;
  };
  _Float16* xh    = (_Float16*)carve((size_t)Mpad*F0*2);
  _Float16* w1h   = (_Float16*)carve((size_t)H1n*F0*2);
  float*    h1    = (float*)   carve((size_t)Mpad*H1n*4);
  float*    mean1 = (float*)   carve((size_t)H1n*4);
  float*    var1  = (float*)   carve((size_t)H1n*4);
  float*    mean2 = (float*)   carve((size_t)H2n*4);
  float*    var2  = (float*)   carve((size_t)H2n*4);
  float*    p1    = (float*)   carve((size_t)Nn*4);
  float*    p2    = (float*)   carve((size_t)Nn*4);
  float*    ev    = (float*)   carve((size_t)Ee*4);
  float*    exv   = (float*)   carve((size_t)Ee*4);
  float*    scv   = (float*)   carve((size_t)Ee*4);
  unsigned* mkey  = (unsigned*)carve((size_t)Nn*4);
  float*    den   = (float*)   carve((size_t)Nn*4);
  unsigned long long* keys = (unsigned long long*)carve((size_t)NPAD*8);
  int*      clus  = (int*)     carve((size_t)Nn*4);
  float*    mul   = (float*)   carve((size_t)Nn*4);
  int*      nb    = (int*)     carve((size_t)Nn*4);
  float*    nx    = (float*)   carve((size_t)Nn*H1n*4);
  _Float16* nxh   = (_Float16*)carve((size_t)Mpad*H1n*2);
  _Float16* w2h   = (_Float16*)carve((size_t)H2n*H1n*2);
  float*    sums  = (float*)   carve((size_t)(Bb+1)*H2n*4);
  float*    cnt   = (float*)   carve((size_t)(Bb+1)*4);
  float*    pooled= (float*)   carve((size_t)Bb*H2n*4);
  float*    hid   = (float*)   carve((size_t)Bb*FC1*4);
  float*    h2    = h1;   // reuse: h1 fully consumed before GEMM2 writes h2

  const int T = 256, GS = 4096;
  const int EB = (Ee + T - 1) / T;

  // ---- layer 1: f16 staging + WMMA GEMM + BN + ReLU ----
  k_fill_f16<<<GS,T,0,stream>>>(xh, 0.f, (size_t)Mpad*F0);
  k_cvt_f16 <<<GS,T,0,stream>>>(x,  xh,  (size_t)Nn*F0);
  k_cvt_f16 <<<GS,T,0,stream>>>(W1, w1h, (size_t)H1n*F0);
  {
    dim3 g(H1n/32, Mpad/32);
    k_wmma_gemm_nt<<<g, 32, 0, stream>>>(xh, w1h, b1, h1, H1n, F0);
  }
  k_colstats<<<H1n, T, 0, stream>>>(h1, Nn, H1n, mean1, var1);
  k_bn_relu <<<GS, T, 0, stream>>>(h1, mean1, var1, g1v, be1, (size_t)Nn*H1n, H1n);

  // ---- edge scores: factored projection + segment softmax over dst ----
  k_node_proj<<<Nn, T, 0, stream>>>(h1, Wp, p1, p2, H1n);
  k_fill_u32 <<<GS, T, 0, stream>>>(mkey, 0u, (size_t)Nn);
  k_fill_f32 <<<GS, T, 0, stream>>>(den, 0.f, (size_t)Nn);
  k_edge_e    <<<EB, T, 0, stream>>>(p1, p2, src, dst, bp, ev, mkey, Ee);
  k_edge_ex   <<<EB, T, 0, stream>>>(ev, mkey, dst, exv, den, Ee);
  k_edge_score<<<EB, T, 0, stream>>>(exv, den, dst, scv, Ee);

  // ---- stable descending sort of scores (bitonic on 64-bit keys) ----
  k_build_keys<<<NPAD/T, T, 0, stream>>>(scv, keys, Ee, NPAD);
  for (int kk = 2; kk <= NPAD; kk <<= 1)
    for (int jj = kk >> 1; jj > 0; jj >>= 1)
      k_bitonic<<<NPAD/T, T, 0, stream>>>(keys, jj, kk);

  // ---- sequential greedy matching + cluster assignment ----
  k_fill_i32<<<GS, T, 0, stream>>>(clus, -1, (size_t)Nn);
  k_fill_f32<<<GS, T, 0, stream>>>(mul, 1.f, (size_t)Nn);
  k_match<<<1, 1, 0, stream>>>(keys, src, dst, scv, clus, mul, Ee, Nn);
  k_fill_i32<<<GS, T, 0, stream>>>(nb, Bb, (size_t)Nn);
  k_scatter_batch<<<(Nn+T-1)/T, T, 0, stream>>>(clus, batch, nb, Nn);

  // ---- new_x = segment_sum(h, cluster) * mult, staged to f16 ----
  k_fill_f32   <<<GS, T, 0, stream>>>(nx, 0.f, (size_t)Nn*H1n);
  k_scatter_add<<<GS, T, 0, stream>>>(h1, clus, nx, (size_t)Nn*H1n, H1n);
  k_fill_f16   <<<GS, T, 0, stream>>>(nxh, 0.f, (size_t)Mpad*H1n);
  k_scale_cvt  <<<GS, T, 0, stream>>>(nx, mul, nxh, (size_t)Nn*H1n, H1n);

  // ---- layer 2: WMMA GEMM + BN + ReLU (h2 aliases h1 region) ----
  k_cvt_f16<<<GS, T, 0, stream>>>(W2, w2h, (size_t)H2n*H1n);
  {
    dim3 g(H2n/32, Mpad/32);
    k_wmma_gemm_nt<<<g, 32, 0, stream>>>(nxh, w2h, b2, h2, H2n, H1n);
  }
  k_colstats<<<H2n, T, 0, stream>>>(h2, Nn, H2n, mean2, var2);
  k_bn_relu <<<GS, T, 0, stream>>>(h2, mean2, var2, g2v, be2, (size_t)Nn*H2n, H2n);

  // ---- batch mean pooling + FC head ----
  k_fill_f32<<<GS, T, 0, stream>>>(sums, 0.f, (size_t)(Bb+1)*H2n);
  k_fill_f32<<<GS, T, 0, stream>>>(cnt, 0.f, (size_t)(Bb+1));
  k_pool   <<<GS, T, 0, stream>>>(h2, nb, sums, cnt, (size_t)Nn*H2n, H2n);
  k_pooled <<<(Bb*H2n + T-1)/T, T, 0, stream>>>(sums, cnt, pooled, Bb, H2n);
  k_fc     <<<(Bb*FC1 + T-1)/T, T, 0, stream>>>(pooled, Wfc, bfc, hid, Bb, FC1, H2n, 1);
  k_fc     <<<(Bb*OUTC + T-1)/T, T, 0, stream>>>(hid, Wfc1, bfc1, out, Bb, OUTC, FC1, 0);
}
